// AggregationMPNN_60979945668613
// MI455X (gfx1250) — matrix operand order repaired and checked by reading, not compiled, not run on hardware
//
#include <hip/hip_runtime.h>
#include <hip/hip_bf16.h>

// ---------------- problem constants ----------------
#define Bn     64
#define Nn     128
#define FNv    64
#define FEv    16
#define OUTD   128
#define S_LEN  1033
#define EMBv   50
#define LHv    8
#define MOMOv  200
#define SIDEv  963
#define FEATS  468   // 128+45+45 + 128+45+45 + 32

typedef __attribute__((ext_vector_type(16))) _Float16 v16h;
typedef __attribute__((ext_vector_type(8)))  float    v8f;

// 16B payload with only 4B alignment requirement (dword-aligned b128 loads ok)
struct f4 { float x, y, z, w; };

// A-fragment K mapping for 16-bit 16x32 A tile (ISA 7.12.2), used in tail path
__device__ __forceinline__ int kmapA(int e, int hi) {
    int v = e >> 1, h = e & 1;
    return ((v & 3) * 2 + h) + ((v >> 2) * 16) + hi * 8;
}

template<int B>
__device__ __forceinline__ void cvt8(v16h& d, const f4& p, const f4& q) {
    d[B + 0] = (_Float16)p.x; d[B + 1] = (_Float16)p.y;
    d[B + 2] = (_Float16)p.z; d[B + 3] = (_Float16)p.w;
    d[B + 4] = (_Float16)q.x; d[B + 5] = (_Float16)q.y;
    d[B + 6] = (_Float16)q.z; d[B + 7] = (_Float16)q.w;
}

// -------------------------------------------------------------------------
// Fused GEMM:  C = act( [A1 | A2] @ Wt^T + bias ), Wt stored [N][K].
// Requirements: M % 64 == 0 (true for every call), K1 % 32 == 0 when HAS_A2.
// One wave per block; wave owns a 64(M) x 16(N) strip = 4 accumulators;
// B fragment loaded once per K-chunk, reused by 4 WMMAs.
// Main loop is branch-free: 4x b128 loads per fragment. K%32 tail guarded.
// -------------------------------------------------------------------------
template<bool HAS_A2, bool HAS_BIAS, bool RELU, bool HAS_MASK, bool HAS_CINIT, bool TRANS_C>
__global__ __launch_bounds__(32)
void gemm_wmma(const float* __restrict__ A1, int lda1, int K1, long sA1,
               const float* __restrict__ A2, int lda2, long sA2,
               const float* __restrict__ Wt, int ldw, long sW,
               const float* __restrict__ bias,
               const float* __restrict__ Cinit, int ldci, long sCi,
               float* __restrict__ C, int ldc, long sC,
               int K, int N,
               const float* __restrict__ mask, long sMask,
               const float* __restrict__ prev, int ldprev)
{
    const int lane = threadIdx.x & 31;
    const int r16  = lane & 15;
    const int hi   = lane >> 4;
    const int col  = blockIdx.x * 16 + r16;
    const int colc = (col < N) ? col : (N - 1);      // clamp; garbage cols not stored
    const int row0 = blockIdx.y * 64;
    const int bz   = blockIdx.z;

    const float* a1 = A1 + (long)bz * sA1;
    const float* a2 = HAS_A2 ? (A2 + (long)bz * sA2) : nullptr;
    const float* wt = Wt + (long)bz * sW + (long)colc * ldw;
    C += (long)bz * sC;

    v8f acc[4];
    #pragma unroll
    for (int t = 0; t < 4; ++t) {
        #pragma unroll
        for (int rr = 0; rr < 8; ++rr) {
            if (HAS_CINIT) {
                int row = row0 + t * 16 + rr + hi * 8;
                acc[t][rr] = Cinit[(long)bz * sCi + (long)row * ldci + colc];
            } else {
                acc[t][rr] = 0.f;
            }
        }
    }

    const int kMain = K & ~31;
    for (int k0 = 0; k0 < kMain; k0 += 32) {
        // ---- B fragment: runs Wt[colc][k0+hi*8,+8) and [k0+16+hi*8,+8)
        const float* bp = wt + k0 + hi * 8;
        f4 b0 = *(const f4*)(bp);
        f4 b1 = *(const f4*)(bp + 4);
        f4 b2 = *(const f4*)(bp + 16);
        f4 b3 = *(const f4*)(bp + 20);
        v16h bfrag;
        cvt8<0>(bfrag, b0, b1);
        cvt8<8>(bfrag, b2, b3);

        // ---- 4 A fragments / 4 WMMAs reusing bfrag
        #pragma unroll
        for (int t = 0; t < 4; ++t) {
            const int row = row0 + t * 16 + r16;
            const float* ap;
            if (HAS_A2 && k0 >= K1) ap = a2 + (long)row * lda2 + (k0 - K1);
            else                    ap = a1 + (long)row * lda1 + k0;
            // lane's A data: runs [k0+hi*8,+8) and [k0+16+hi*8,+8)
            f4 p0 = *(const f4*)(ap + hi * 8);
            f4 p1 = *(const f4*)(ap + hi * 8 + 4);
            f4 p2 = *(const f4*)(ap + 16 + hi * 8);
            f4 p3 = *(const f4*)(ap + 16 + hi * 8 + 4);
            v16h afrag;
            cvt8<0>(afrag, p0, p1);
            cvt8<8>(afrag, p2, p3);
            acc[t] = __builtin_amdgcn_wmma_f32_16x16x32_f16(
                         false, afrag, false, bfrag, (short)0, acc[t], false, false);
        }
    }

    // ---- guarded tail chunk (only for K % 32 != 0)
    if (kMain < K) {
        const int k0 = kMain;
        v16h bfrag;
        #pragma unroll
        for (int e = 0; e < 16; ++e) {
            int k = k0 + kmapA(e, hi);
            bfrag[e] = (_Float16)((k < K) ? wt[k] : 0.f);
        }
        #pragma unroll
        for (int t = 0; t < 4; ++t) {
            const int row = row0 + t * 16 + r16;
            v16h afrag;
            #pragma unroll
            for (int e = 0; e < 16; ++e) {
                int k = k0 + kmapA(e, hi);
                float v = 0.f;
                if (k < K) {
                    if (HAS_A2 && k >= K1) v = a2[(long)row * lda2 + (k - K1)];
                    else                   v = a1[(long)row * lda1 + k];
                }
                afrag[e] = (_Float16)v;
            }
            acc[t] = __builtin_amdgcn_wmma_f32_16x16x32_f16(
                         false, afrag, false, bfrag, (short)0, acc[t], false, false);
        }
    }

    // ---- epilogue: bias, relu, masked row-select, store
    float bv = HAS_BIAS ? bias[colc] : 0.f;
    #pragma unroll
    for (int t = 0; t < 4; ++t) {
        #pragma unroll
        for (int rr = 0; rr < 8; ++rr) {
            const int row = row0 + t * 16 + rr + hi * 8;
            float v = acc[t][rr] + bv;
            if (RELU) v = fmaxf(v, 0.f);
            if (HAS_MASK) {
                if (mask[(long)bz * sMask + row] == 0.f)
                    v = prev[(long)row * ldprev + colc];
            }
            if (col < N) {
                if (TRANS_C) C[(long)col * ldc + row] = v;
                else         C[(long)row * ldc + col] = v;
            }
        }
    }
}

// -------------------------------------------------------------------------
// Weight transpose prep: src [R][Cc] -> dst [Cc][R]
// -------------------------------------------------------------------------
__global__ void transpose_kernel(const float* __restrict__ src, float* __restrict__ dst,
                                 int R, int Cc)
{
    const int c = blockIdx.x * 16 + threadIdx.x;
    const int r = blockIdx.y * 16 + threadIdx.y;
    if (r < R && c < Cc) dst[(long)c * R + r] = src[(long)r * Cc + c];
}

// -------------------------------------------------------------------------
// Edge aggregation: agg_e[b,n,f] = sum_m adj[b,n,m]*edges[b,n,m,f], plus
// node mask.  Fully-coalesced: lane streams whole 64B edge rows (f4 x4),
// keeps a 16-wide accumulator, wave butterfly-reduces.  Streams 134MB once.
// -------------------------------------------------------------------------
__global__ __launch_bounds__(32)
void edge_agg_kernel(const float* __restrict__ adj, const float* __restrict__ edges,
                     float* __restrict__ agg_e, float* __restrict__ nmask)
{
    const int n = blockIdx.x, b = blockIdx.y, lane = threadIdx.x;
    const float* adjRow = adj + ((long)b * Nn + n) * Nn;
    const float* eRow   = edges + ((long)b * Nn + n) * Nn * FEv;

    float s[FEv];
    #pragma unroll
    for (int f = 0; f < FEv; ++f) s[f] = 0.f;
    float deg = 0.f;

    for (int m = lane; m < Nn; m += 32) {
        const float a = adjRow[m];
        deg += a;
        const f4* ep = (const f4*)(eRow + (long)m * FEv);
        f4 e0 = ep[0], e1 = ep[1], e2 = ep[2], e3 = ep[3];
        s[0]  += a * e0.x; s[1]  += a * e0.y; s[2]  += a * e0.z; s[3]  += a * e0.w;
        s[4]  += a * e1.x; s[5]  += a * e1.y; s[6]  += a * e1.z; s[7]  += a * e1.w;
        s[8]  += a * e2.x; s[9]  += a * e2.y; s[10] += a * e2.z; s[11] += a * e2.w;
        s[12] += a * e3.x; s[13] += a * e3.y; s[14] += a * e3.z; s[15] += a * e3.w;
    }
    #pragma unroll
    for (int o = 16; o > 0; o >>= 1) {
        deg += __shfl_xor(deg, o, 32);
        #pragma unroll
        for (int f = 0; f < FEv; ++f) s[f] += __shfl_xor(s[f], o, 32);
    }
    if (lane == 0) {
        float* dst = agg_e + ((long)b * Nn + n) * FEv;
        #pragma unroll
        for (int f = 0; f < FEv; ++f) dst[f] = s[f];
        nmask[(long)b * Nn + n] = (deg > 0.f) ? 1.f : 0.f;
    }
}

// -------------------------------------------------------------------------
// Masked readout sum: out[b,k] = sum_n mask[b,n]*r[b,n,k] -> feats
// -------------------------------------------------------------------------
__global__ void readout_sum_kernel(const float* __restrict__ r, const float* __restrict__ nmask,
                                   float* __restrict__ feats, int featOff)
{
    const int k = blockIdx.x * blockDim.x + threadIdx.x;
    const int b = blockIdx.y;
    if (k >= OUTD) return;
    const float* rb = r + (long)b * Nn * OUTD;
    const float* mb = nmask + (long)b * Nn;
    float s = 0.f;
    for (int n = 0; n < Nn; ++n) s += mb[n] * rb[(long)n * OUTD + k];
    feats[(long)b * FEATS + featOff + k] = s;
}

// -------------------------------------------------------------------------
// Embedding + LayerNorm over EMB=50, one wave per (b,s) row.
// -------------------------------------------------------------------------
__global__ __launch_bounds__(32)
void embed_ln_kernel(const int* __restrict__ ids,
                     const float* __restrict__ wemb, const float* __restrict__ pemb,
                     const float* __restrict__ gamma, const float* __restrict__ beta,
                     float* __restrict__ x)
{
    const int s = blockIdx.x, b = blockIdx.y, lane = threadIdx.x;
    const int id = ids[(long)b * S_LEN + s];
    const int e0 = lane, e1 = lane + 32;
    float v0 = 0.f, v1 = 0.f;
    if (e0 < EMBv) v0 = wemb[(long)id * EMBv + e0] + pemb[(long)s * EMBv + e0];
    if (e1 < EMBv) v1 = wemb[(long)id * EMBv + e1] + pemb[(long)s * EMBv + e1];
    float sum = v0 + v1;
    #pragma unroll
    for (int o = 16; o > 0; o >>= 1) sum += __shfl_xor(sum, o, 32);
    const float mean = sum / (float)EMBv;
    float d0 = (e0 < EMBv) ? v0 - mean : 0.f;
    float d1 = (e1 < EMBv) ? v1 - mean : 0.f;
    float sq = d0 * d0 + d1 * d1;
    #pragma unroll
    for (int o = 16; o > 0; o >>= 1) sq += __shfl_xor(sq, o, 32);
    const float rstd = rsqrtf(sq / (float)EMBv + 1e-12f);
    const long base = ((long)b * S_LEN + s) * EMBv;
    if (e0 < EMBv) x[base + e0] = gamma[e0] * d0 * rstd + beta[e0];
    if (e1 < EMBv) x[base + e1] = gamma[e1] * d1 * rstd + beta[e1];
}

// -------------------------------------------------------------------------
// LSTM recurrence: one wave per batch row, 1033 sequential steps.
// Thread j owns gate unit j (0..31); LDS carries h[8], c[8].
// -------------------------------------------------------------------------
__global__ __launch_bounds__(32)
void lstm_scan_kernel(const float* __restrict__ xW, const float* __restrict__ W_hh,
                      const float* __restrict__ b_hh, float* __restrict__ hs)
{
    const int b = blockIdx.x, j = threadIdx.x;
    __shared__ float hsh[LHv], csh[LHv], gsh[4 * LHv];
    if (j < LHv) { hsh[j] = 0.f; csh[j] = 0.f; }
    float wrow[LHv];
    #pragma unroll
    for (int k = 0; k < LHv; ++k) wrow[k] = W_hh[j * LHv + k];
    const float bj = b_hh[j];
    __syncthreads();
    const float* xWb = xW + (long)b * S_LEN * 32;
    float* hsb = hs + (long)b * S_LEN * LHv;
    for (int s = 0; s < S_LEN; ++s) {
        float g = xWb[(long)s * 32 + j] + bj;
        #pragma unroll
        for (int k = 0; k < LHv; ++k) g += hsh[k] * wrow[k];
        gsh[j] = g;
        __syncthreads();
        if (j < LHv) {
            float ig = 1.f / (1.f + __expf(-gsh[j]));
            float fg = 1.f / (1.f + __expf(-gsh[LHv + j]));
            float gg = tanhf(gsh[2 * LHv + j]);
            float og = 1.f / (1.f + __expf(-gsh[3 * LHv + j]));
            float c  = fg * csh[j] + ig * gg;
            float h  = og * tanhf(c);
            csh[j] = c; hsh[j] = h;
            hsb[(long)s * LHv + j] = h;
        }
        __syncthreads();
    }
}

// -------------------------------------------------------------------------
// BatchNorm1d (training mode, batch stats over 64 rows), one thread per col.
// -------------------------------------------------------------------------
__global__ void batchnorm_kernel(const float* __restrict__ p, const float* __restrict__ gamma,
                                 const float* __restrict__ beta, float* __restrict__ out)
{
    const int c = blockIdx.x * blockDim.x + threadIdx.x;
    if (c >= 256) return;
    float s = 0.f, s2 = 0.f;
    for (int r = 0; r < Bn; ++r) { float v = p[(long)r * 256 + c]; s += v; s2 += v * v; }
    const float m  = s / (float)Bn;
    const float vv = s2 / (float)Bn - m * m;
    const float rs = rsqrtf(vv + 1e-5f);
    for (int r = 0; r < Bn; ++r)
        out[(long)r * 256 + c] = gamma[c] * (p[(long)r * 256 + c] - m) * rs + beta[c];
}

// -------------------------------------------------------------------------
// Final dot: out[b] = feats[b,:] @ W_final + b_final
// -------------------------------------------------------------------------
__global__ void final_dot_kernel(const float* __restrict__ feats, const float* __restrict__ Wf,
                                 const float* __restrict__ bf, float* __restrict__ out)
{
    const int b = blockIdx.x * blockDim.x + threadIdx.x;
    if (b >= Bn) return;
    float s = bf[0];
    for (int j = 0; j < FEATS; ++j) s += feats[(long)b * FEATS + j] * Wf[j];
    out[b] = s;
}

// =========================================================================
extern "C" void kernel_launch(void* const* d_in, const int* in_sizes, int n_in,
                              void* d_out, int out_size, void* d_ws, size_t ws_size,
                              hipStream_t stream)
{
    (void)in_sizes; (void)n_in; (void)out_size; (void)ws_size;

    // ---- inputs (setup_inputs dict order) ----
    const float* adjG[2]   = { (const float*)d_in[0], (const float*)d_in[3] };
    const float* nodesG[2] = { (const float*)d_in[1], (const float*)d_in[4] };
    const float* edgesG[2] = { (const float*)d_in[2], (const float*)d_in[5] };
    const float* momoF[2]  = { (const float*)d_in[6], (const float*)d_in[7] };
    const int*   protIds[2]= { (const int*)d_in[8],   (const int*)d_in[9]  };
    const float* side_eff  = (const float*)d_in[10];
    const float* wemb      = (const float*)d_in[11];
    const float* pemb      = (const float*)d_in[12];
    const float* ln_g      = (const float*)d_in[13];
    const float* ln_b      = (const float*)d_in[14];
    const float* W_ih      = (const float*)d_in[15];   // [32,50] == Wt layout already
    const float* W_hh      = (const float*)d_in[16];   // [32,8]
    const float* b_ih      = (const float*)d_in[17];
    const float* b_hh      = (const float*)d_in[18];
    const float* dec_W1    = (const float*)d_in[19];   // [8264,256]
    const float* dec_b1    = (const float*)d_in[20];
    const float* bn_g      = (const float*)d_in[21];
    const float* bn_b      = (const float*)d_in[22];
    const float* dec_W2    = (const float*)d_in[23];   // [256,45]
    const float* dec_b2    = (const float*)d_in[24];
    const float* W_msgG[2]  = { (const float*)d_in[25], (const float*)d_in[31] };
    const float* W_edgeG[2] = { (const float*)d_in[26], (const float*)d_in[32] };
    const float* W_upG[2]   = { (const float*)d_in[27], (const float*)d_in[33] };
    const float* b_upG[2]   = { (const float*)d_in[28], (const float*)d_in[34] };
    const float* W_readG[2] = { (const float*)d_in[29], (const float*)d_in[35] };
    const float* b_readG[2] = { (const float*)d_in[30], (const float*)d_in[36] };
    const float* W_momo    = (const float*)d_in[37];   // [200,45]
    const float* b_momo    = (const float*)d_in[38];
    const float* W_side    = (const float*)d_in[39];   // [963,32]
    const float* b_side    = (const float*)d_in[40];
    const float* W_final   = (const float*)d_in[41];   // [468,1]
    const float* b_final   = (const float*)d_in[42];

    // ---- workspace carve (floats) ----
    float* w = (float*)d_ws;
    size_t off = 0;
    auto alloc = [&](size_t n) { float* p = w + off; off += n; return p; };
    float* agg_e    = alloc((size_t)Bn * Nn * FEv);
    float* nmask    = alloc((size_t)Bn * Nn);
    float* edge_msg = alloc((size_t)Bn * Nn * FNv);
    float* hmT      = alloc((size_t)Bn * FNv * Nn);     // [b][feat][node]
    float* msg      = alloc((size_t)Bn * Nn * FNv);
    float* hA       = alloc((size_t)Bn * Nn * FNv);
    float* hB       = alloc((size_t)Bn * Nn * FNv);
    float* rbuf     = alloc((size_t)Bn * Nn * OUTD);
    float* x_ln     = alloc((size_t)Bn * S_LEN * EMBv);
    float* xW       = alloc((size_t)Bn * S_LEN * 32);
    float* hsbuf    = alloc((size_t)Bn * S_LEN * LHv);
    float* pbuf     = alloc((size_t)Bn * 256);
    float* pbn      = alloc((size_t)Bn * 256);
    float* feats    = alloc((size_t)Bn * FEATS);
    // transposed weights [N][K]
    float* WtEdge   = alloc((size_t)FNv * FEv);
    float* WtMsg    = alloc((size_t)FNv * FNv);
    float* WtUp     = alloc((size_t)FNv * 2 * FNv);
    float* WtRead   = alloc((size_t)OUTD * 2 * FNv);
    float* WtDec1   = alloc((size_t)256 * S_LEN * LHv);
    float* WtDec2   = alloc((size_t)45 * 256 + 3);
    float* WtMomo   = alloc((size_t)45 * MOMOv);
    float* WtSide   = alloc((size_t)32 * SIDEv);

    auto transpose = [&](const float* src, float* dst, int R, int Cc) {
        dim3 grid((Cc + 15) / 16, (R + 15) / 16);
        transpose_kernel<<<grid, dim3(16, 16), 0, stream>>>(src, dst, R, Cc);
    };

    #define LAUNCH_GEMM(A2F, BIASF, RELUF, MASKF, CINITF, TRANSCF,                         \
                        A1, lda1, K1, sA1, A2, lda2, sA2, Wt, ldw, sW, bias,               \
                        Cinit, ldci, sCi, Cp, ldc, sC, M, K, N, mask, sMask, prev, ldprev, \
                        batch)                                                             \
        gemm_wmma<A2F, BIASF, RELUF, MASKF, CINITF, TRANSCF>                               \
            <<<dim3(((N) + 15) / 16, (M) / 64, (batch)), dim3(32), 0, stream>>>(           \
                A1, lda1, K1, sA1, A2, lda2, sA2, Wt, ldw, sW, bias,                       \
                Cinit, ldci, sCi, Cp, ldc, sC, K, N, mask, sMask, prev, ldprev)

    const int featOut[2]  = { 0, 218 };
    const int featP[2]    = { 128, 346 };
    const int featMomo[2] = { 173, 391 };

    // ==================== MPNN branches ====================
    for (int g = 0; g < 2; ++g) {
        transpose(W_edgeG[g], WtEdge, FEv, FNv);
        transpose(W_msgG[g],  WtMsg,  FNv, FNv);
        transpose(W_upG[g],   WtUp,   2 * FNv, FNv);
        transpose(W_readG[g], WtRead, 2 * FNv, OUTD);

        edge_agg_kernel<<<dim3(Nn, Bn), dim3(32), 0, stream>>>(adjG[g], edgesG[g], agg_e, nmask);

        // edge_msg = agg_e @ W_edge   [8192,16]x[16,64]
        LAUNCH_GEMM(false, false, false, false, false, false,
                    agg_e, FEv, 0, 0L, nullptr, 0, 0L, WtEdge, FEv, 0L, nullptr,
                    nullptr, 0, 0L, edge_msg, FNv, 0L, Bn * Nn, FEv, FNv,
                    nullptr, 0L, nullptr, 0, 1);

        const float* hcur = nodesG[g];
        float* houts[3] = { hA, hB, hA };
        for (int t = 0; t < 3; ++t) {
            // hmT[b] = (h[b] @ W_msg)^T  : batched, TRANS_C output [feat][node]
            LAUNCH_GEMM(false, false, false, false, false, true,
                        hcur, FNv, 0, (long)Nn * FNv, nullptr, 0, 0L, WtMsg, FNv, 0L, nullptr,
                        nullptr, 0, 0L, hmT, Nn, (long)FNv * Nn, Nn, FNv, FNv,
                        nullptr, 0L, nullptr, 0, Bn);
            // msg[b] = adj[b] @ hm[b] + edge_msg[b]   (hmT is Wt[N=feat][K=node])
            LAUNCH_GEMM(false, false, false, false, true, false,
                        adjG[g], Nn, 0, (long)Nn * Nn, nullptr, 0, 0L, hmT, Nn, (long)FNv * Nn,
                        nullptr, edge_msg, FNv, (long)Nn * FNv,
                        msg, FNv, (long)Nn * FNv, Nn, Nn, FNv,
                        nullptr, 0L, nullptr, 0, Bn);
            // h_next = mask ? relu([h|msg]@W_up + b_up) : h
            LAUNCH_GEMM(true, true, true, true, false, false,
                        hcur, FNv, FNv, 0L, msg, FNv, 0L, WtUp, 2 * FNv, 0L, b_upG[g],
                        nullptr, 0, 0L, houts[t], FNv, 0L, Bn * Nn, 2 * FNv, FNv,
                        nmask, 0L, hcur, FNv, 1);
            hcur = houts[t];
        }
        // r = relu([h|nodes]@W_read + b_read)   [8192,128]x[128,128]
        LAUNCH_GEMM(true, true, true, false, false, false,
                    hcur, FNv, FNv, 0L, nodesG[g], FNv, 0L, WtRead, 2 * FNv, 0L, b_readG[g],
                    nullptr, 0, 0L, rbuf, OUTD, 0L, Bn * Nn, 2 * FNv, OUTD,
                    nullptr, 0L, nullptr, 0, 1);
        readout_sum_kernel<<<dim3(1, Bn), dim3(128), 0, stream>>>(rbuf, nmask, feats, featOut[g]);
    }

    // ==================== Protein branches ====================
    transpose(dec_W1, WtDec1, S_LEN * LHv, 256);
    transpose(dec_W2, WtDec2, 256, 45);
    transpose(W_momo, WtMomo, MOMOv, 45);
    transpose(W_side, WtSide, SIDEv, 32);

    for (int d = 0; d < 2; ++d) {
        embed_ln_kernel<<<dim3(S_LEN, Bn), dim3(32), 0, stream>>>(
            protIds[d], wemb, pemb, ln_g, ln_b, x_ln);
        // xW = x_ln @ W_ih^T + b_ih ; W_ih already [N=32][K=50]
        LAUNCH_GEMM(false, true, false, false, false, false,
                    x_ln, EMBv, 0, 0L, nullptr, 0, 0L, W_ih, EMBv, 0L, b_ih,
                    nullptr, 0, 0L, xW, 32, 0L, Bn * S_LEN, EMBv, 32,
                    nullptr, 0L, nullptr, 0, 1);
        lstm_scan_kernel<<<dim3(Bn), dim3(32), 0, stream>>>(xW, W_hh, b_hh, hsbuf);
        // p = relu(hs_flat @ dec_W1 + dec_b1)   [64,8264]x[8264,256]
        LAUNCH_GEMM(false, true, true, false, false, false,
                    hsbuf, S_LEN * LHv, 0, 0L, nullptr, 0, 0L, WtDec1, S_LEN * LHv, 0L, dec_b1,
                    nullptr, 0, 0L, pbuf, 256, 0L, Bn, S_LEN * LHv, 256,
                    nullptr, 0L, nullptr, 0, 1);
        batchnorm_kernel<<<dim3(1), dim3(256), 0, stream>>>(pbuf, bn_g, bn_b, pbn);
        // p45 = pbn @ dec_W2 + dec_b2 -> feats
        LAUNCH_GEMM(false, true, false, false, false, false,
                    pbn, 256, 0, 0L, nullptr, 0, 0L, WtDec2, 256, 0L, dec_b2,
                    nullptr, 0, 0L, feats + featP[d], FEATS, 0L, Bn, 256, 45,
                    nullptr, 0L, nullptr, 0, 1);
        // momo = momo_fts @ W_momo + b_momo -> feats
        LAUNCH_GEMM(false, true, false, false, false, false,
                    momoF[d], MOMOv, 0, 0L, nullptr, 0, 0L, WtMomo, MOMOv, 0L, b_momo,
                    nullptr, 0, 0L, feats + featMomo[d], FEATS, 0L, Bn, MOMOv, 45,
                    nullptr, 0L, nullptr, 0, 1);
    }

    // side = side_effect @ W_side + b_side -> feats
    LAUNCH_GEMM(false, true, false, false, false, false,
                side_eff, SIDEv, 0, 0L, nullptr, 0, 0L, WtSide, SIDEv, 0L, b_side,
                nullptr, 0, 0L, feats + 436, FEATS, 0L, Bn, SIDEv, 32,
                nullptr, 0L, nullptr, 0, 1);

    final_dot_kernel<<<dim3(1), dim3(64), 0, stream>>>(feats, W_final, b_final, (float*)d_out);
}
